// RecurrentGCN_48825188221182
// MI455X (gfx1250) — matrix operand order repaired and checked by reading, not compiled, not run on hardware
//
#include <hip/hip_runtime.h>

#define BB   4
#define NN   20000
#define EE   320000
#define DINN 64
#define DM   64
#define DOO  32
#define RTOT (BB * NN)        /* 80000 rows */
#define R64  (RTOT * DM)      /* 5,120,000 floats */

typedef __attribute__((ext_vector_type(16))) __bf16 v16bf;
typedef __attribute__((ext_vector_type(8)))  float  v8f;

// ---------------------------------------------------------------- degree
__global__ void deg_kernel(const int* __restrict__ dst, const float* __restrict__ w,
                           float* __restrict__ deg) {
    int e = blockIdx.x * blockDim.x + threadIdx.x;
    if (e < EE) unsafeAtomicAdd(&deg[dst[e]], w[e]);
}

__device__ __forceinline__ float dinvf(float d) {
    return d > 0.f ? rsqrtf(fmaxf(d, 1e-12f)) : 0.f;
}

__global__ void norm_kernel(const int* __restrict__ src, const int* __restrict__ dst,
                            const float* __restrict__ w, const float* __restrict__ deg,
                            float* __restrict__ nrm) {
    int e = blockIdx.x * blockDim.x + threadIdx.x;
    if (e < EE) nrm[e] = dinvf(deg[src[e]]) * w[e] * dinvf(deg[dst[e]]);
}

// ---------------------------------------------------------------- X @ W (WMMA bf16)
// LDS holds W pre-swizzled into per-lane-contiguous B fragments:
//   sW[((kt*4 + nt)*32 + lane)*16 + j] = bf16( W[kt*32 + (lane>>4)*8 + (j&7) + (j>>3)*16][nt*16 + (lane&15)] )
// so each lane's fragment is one aligned 32-byte v16bf load (2x ds_load_b128).
__global__ void xw_gemm(const float* __restrict__ X, const float* __restrict__ W,
                        float* __restrict__ out) {
    __shared__ alignas(32) __bf16 sW[DINN * DM];   // 4096 halfs = 8 KB
    for (int idx = threadIdx.x; idx < DINN * DM; idx += blockDim.x) {
        int j    = idx & 15;
        int lane = (idx >> 4) & 31;
        int nt   = (idx >> 9) & 3;
        int kt   = idx >> 11;
        int row  = kt * 32 + (lane >> 4) * 8 + (j & 7) + (j >> 3) * 16;
        int col  = nt * 16 + (lane & 15);
        sW[idx]  = (__bf16)W[row * DM + col];
    }
    __syncthreads();
    const v16bf* sWf = (const v16bf*)sW;

    const int wave  = threadIdx.x >> 5;
    const int lane  = threadIdx.x & 31;
    const int chunk = blockIdx.x * (blockDim.x >> 5) + wave;
    const int row0  = chunk * 16;
    if (row0 >= RTOT) return;
    const int half = lane >> 4;
    const int mn   = lane & 15;                    // row for A-load, col for D

    v8f acc[4] = {};
    #pragma unroll
    for (int kt = 0; kt < 2; ++kt) {
        v16bf a;
        const float* p = X + (size_t)(row0 + mn) * DINN + kt * 32;
        #pragma unroll
        for (int j = 0; j < 8; ++j) a[j]     = (__bf16)p[half * 8 + j];
        #pragma unroll
        for (int j = 0; j < 8; ++j) a[8 + j] = (__bf16)p[16 + half * 8 + j];
        #pragma unroll
        for (int nt = 0; nt < 4; ++nt) {
            v16bf bf = sWf[(kt * 4 + nt) * 32 + lane];
            acc[nt] = __builtin_amdgcn_wmma_f32_16x16x32_bf16(
                false, a, false, bf, (short)0, acc[nt], false, false);
        }
    }
    #pragma unroll
    for (int nt = 0; nt < 4; ++nt)
        #pragma unroll
        for (int i = 0; i < 8; ++i)
            out[(size_t)(row0 + i + 8 * half) * DM + nt * 16 + mn] = acc[nt][i];
}

// ---------------------------------------------------------------- edge scatter
// agg[b, dst, :] += norm * xw[b, src, :]   (64 threads per edge, 4 batches)
__global__ void scatter_kernel(const float* __restrict__ xw, const int* __restrict__ src,
                               const int* __restrict__ dst, const float* __restrict__ nrm,
                               float* __restrict__ agg) {
    int e = blockIdx.x * (blockDim.x >> 6) + (threadIdx.x >> 6);
    int d = threadIdx.x & 63;
    if (e >= EE) return;
    int s = src[e], t = dst[e];
    float nv = nrm[e];
    const float* xs = xw  + (size_t)s * DM + d;
    float*       ad = agg + (size_t)t * DM + d;
    #pragma unroll
    for (int b = 0; b < BB; ++b)
        unsafeAtomicAdd(ad + (size_t)b * NN * DM, nv * xs[(size_t)b * NN * DM]);
}

// ---------------------------------------------------------------- GRU gate GEMM
// MODE 0/1: out = sigmoid([agg+bc | H] @ L + lb)
// MODE 2:   Ht = tanh([agg+bc | H*R] @ L + lb);  out = Z*H + (1-Z)*Ht
template <int MODE>
__global__ void gate_gemm(const float* __restrict__ agg, const float* __restrict__ bc,
                          const float* __restrict__ H,   const float* __restrict__ Rb,
                          const float* __restrict__ Zb,  const float* __restrict__ L,
                          const float* __restrict__ lb,  float* __restrict__ out) {
    __shared__ alignas(32) __bf16 sL[2 * DM * DM]; // 8192 halfs = 16 KB, frag-major
    for (int idx = threadIdx.x; idx < 2 * DM * DM; idx += blockDim.x) {
        int j    = idx & 15;
        int lane = (idx >> 4) & 31;
        int nt   = (idx >> 9) & 3;
        int kt   = idx >> 11;                      // 0..3
        int row  = kt * 32 + (lane >> 4) * 8 + (j & 7) + (j >> 3) * 16;
        int col  = nt * 16 + (lane & 15);
        sL[idx]  = (__bf16)L[row * DM + col];
    }
    __syncthreads();
    const v16bf* sLf = (const v16bf*)sL;

    const int wave  = threadIdx.x >> 5;
    const int lane  = threadIdx.x & 31;
    const int chunk = blockIdx.x * (blockDim.x >> 5) + wave;
    const int row0  = chunk * 16;
    if (row0 >= RTOT) return;
    const int half = lane >> 4;
    const int mn   = lane & 15;

    v8f acc[4] = {};
    #pragma unroll
    for (int kt = 0; kt < 4; ++kt) {
        v16bf a;
        if (kt < 2) {
            const float* p  = agg + (size_t)(row0 + mn) * DM + kt * 32;
            const float* bp = bc + kt * 32;
            #pragma unroll
            for (int j = 0; j < 8; ++j) { int o = half * 8 + j;      a[j]     = (__bf16)(p[o] + bp[o]); }
            #pragma unroll
            for (int j = 0; j < 8; ++j) { int o = 16 + half * 8 + j; a[8 + j] = (__bf16)(p[o] + bp[o]); }
        } else {
            const int k2 = (kt - 2) * 32;
            const float* p = H + (size_t)(row0 + mn) * DM + k2;
            if (MODE == 2) {
                const float* pr = Rb + (size_t)(row0 + mn) * DM + k2;
                #pragma unroll
                for (int j = 0; j < 8; ++j) { int o = half * 8 + j;      a[j]     = (__bf16)(p[o] * pr[o]); }
                #pragma unroll
                for (int j = 0; j < 8; ++j) { int o = 16 + half * 8 + j; a[8 + j] = (__bf16)(p[o] * pr[o]); }
            } else {
                #pragma unroll
                for (int j = 0; j < 8; ++j) a[j]     = (__bf16)p[half * 8 + j];
                #pragma unroll
                for (int j = 0; j < 8; ++j) a[8 + j] = (__bf16)p[16 + half * 8 + j];
            }
        }
        #pragma unroll
        for (int nt = 0; nt < 4; ++nt) {
            v16bf bf = sLf[(kt * 4 + nt) * 32 + lane];
            acc[nt] = __builtin_amdgcn_wmma_f32_16x16x32_bf16(
                false, a, false, bf, (short)0, acc[nt], false, false);
        }
    }
    #pragma unroll
    for (int nt = 0; nt < 4; ++nt) {
        int col = nt * 16 + mn;
        float bias = lb[col];
        #pragma unroll
        for (int i = 0; i < 8; ++i) {
            size_t idx = (size_t)(row0 + i + 8 * half) * DM + col;
            float v = acc[nt][i] + bias;
            if (MODE == 2) {
                float z = Zb[idx], h = H[idx];
                out[idx] = z * h + (1.f - z) * tanhf(v);
            } else {
                out[idx] = 1.f / (1.f + expf(-v));
            }
        }
    }
}

// ---------------------------------------------------------------- attention gate
__global__ void attn_gate_kernel(const float* __restrict__ H, const float* __restrict__ gw,
                                 const float* __restrict__ gb, float* __restrict__ gate) {
    int wid  = (blockIdx.x * blockDim.x + threadIdx.x) >> 5;
    int lane = threadIdx.x & 31;
    if (wid >= RTOT) return;
    const float* h = H + (size_t)wid * DM;
    float v = h[lane] * gw[lane] + h[32 + lane] * gw[32 + lane];
    #pragma unroll
    for (int off = 16; off > 0; off >>= 1) v += __shfl_down(v, off, 32);
    if (lane == 0) gate[wid] = v + gb[0];
}

__global__ void softmax_stats(const float* __restrict__ gate, float* __restrict__ gstat) {
    __shared__ float red[256];
    int b = blockIdx.x, t = threadIdx.x;
    const float* g = gate + (size_t)b * NN;
    float mx = -1e30f;
    for (int n = t; n < NN; n += 256) mx = fmaxf(mx, g[n]);
    red[t] = mx; __syncthreads();
    for (int s = 128; s > 0; s >>= 1) { if (t < s) red[t] = fmaxf(red[t], red[t + s]); __syncthreads(); }
    float m = red[0]; __syncthreads();
    float sm = 0.f;
    for (int n = t; n < NN; n += 256) sm += expf(g[n] - m);
    red[t] = sm; __syncthreads();
    for (int s = 128; s > 0; s >>= 1) { if (t < s) red[t] += red[t + s]; __syncthreads(); }
    if (t == 0) { gstat[2 * b] = m; gstat[2 * b + 1] = red[0]; }
}

__global__ void pool_kernel(const float* __restrict__ H, const float* __restrict__ gate,
                            const float* __restrict__ gstat, float* __restrict__ ypool) {
    __shared__ float red[256];
    int b = blockIdx.x, t = threadIdx.x;
    int d = t & 63, s = t >> 6;
    float m = gstat[2 * b], sum = gstat[2 * b + 1];
    float acc = 0.f;
    for (int n = s; n < NN; n += 4) {
        float w = expf(gate[(size_t)b * NN + n] - m);
        acc += w * H[((size_t)b * NN + n) * DM + d];
    }
    red[t] = acc; __syncthreads();
    if (s == 0) ypool[b * DM + d] = (red[d] + red[64 + d] + red[128 + d] + red[192 + d]) / sum;
}

__global__ void head_kernel(const float* __restrict__ ypool, const float* __restrict__ W1,
                            const float* __restrict__ b1, const float* __restrict__ W2,
                            const float* __restrict__ b2, float* __restrict__ y) {
    __shared__ float h1[BB * DM];
    int t = threadIdx.x;
    int b = t >> 6, d = t & 63;
    float a = b1[d];
    for (int k = 0; k < DM; ++k) a += ypool[b * DM + k] * W1[k * DM + d];
    h1[b * DM + d] = fmaxf(a, 0.f);
    __syncthreads();
    if (t < BB * DOO) {
        int bb = t >> 5, o = t & 31;
        float a2 = b2[o];
        for (int k = 0; k < DM; ++k) a2 += h1[bb * DM + k] * W2[k * DOO + o];
        y[bb * DOO + o] = a2;
    }
}

// ---------------------------------------------------------------- launch
extern "C" void kernel_launch(void* const* d_in, const int* in_sizes, int n_in,
                              void* d_out, int out_size, void* d_ws, size_t ws_size,
                              hipStream_t stream) {
    const float* x    = (const float*)d_in[0];
    const int*   ei   = (const int*)d_in[1];
    const float* ew   = (const float*)d_in[2];
    const float* Hprev= (const float*)d_in[3];
    const float* Wg[3]= {(const float*)d_in[4], (const float*)d_in[6], (const float*)d_in[8]};
    const float* bg[3]= {(const float*)d_in[5], (const float*)d_in[7], (const float*)d_in[9]};
    const float* Lg[3]= {(const float*)d_in[10], (const float*)d_in[12], (const float*)d_in[14]};
    const float* lg[3]= {(const float*)d_in[11], (const float*)d_in[13], (const float*)d_in[15]};
    const float* gw   = (const float*)d_in[16];
    const float* gb   = (const float*)d_in[17];
    const float* W1   = (const float*)d_in[18];
    const float* b1   = (const float*)d_in[19];
    const float* W2   = (const float*)d_in[20];
    const float* b2   = (const float*)d_in[21];

    const int* srcI = ei;
    const int* dstI = ei + EE;

    float* y_out = (float*)d_out;              // (4,32)
    float* Hnew  = y_out + BB * DOO;           // (4,20000,64)

    float* ws    = (float*)d_ws;
    float* deg   = ws;                 // 20480 (N rounded)
    float* nrm   = deg  + 20480;       // E
    float* xw    = nrm  + EE;          // R64  (reused per gate)
    float* agg   = xw   + R64;         // R64  (reused per gate)
    float* Zb    = agg  + R64;         // R64
    float* Rb    = Zb   + R64;         // R64
    float* gate  = Rb   + R64;         // RTOT
    float* gstat = gate + RTOT;        // 8
    float* ypool = gstat + 8;          // 256

    const int gemmBlocks = (RTOT / 16 + 7) / 8;          // 625 (8 waves/block)

    hipMemsetAsync(deg, 0, NN * sizeof(float), stream);
    deg_kernel<<<(EE + 255) / 256, 256, 0, stream>>>(dstI, ew, deg);
    norm_kernel<<<(EE + 255) / 256, 256, 0, stream>>>(srcI, dstI, ew, deg, nrm);

    // ---- gate z ----
    hipMemsetAsync(agg, 0, (size_t)R64 * sizeof(float), stream);
    xw_gemm<<<gemmBlocks, 256, 0, stream>>>(x, Wg[0], xw);
    scatter_kernel<<<EE / 4, 256, 0, stream>>>(xw, srcI, dstI, nrm, agg);
    gate_gemm<0><<<gemmBlocks, 256, 0, stream>>>(agg, bg[0], Hprev, nullptr, nullptr,
                                                 Lg[0], lg[0], Zb);
    // ---- gate r ----
    hipMemsetAsync(agg, 0, (size_t)R64 * sizeof(float), stream);
    xw_gemm<<<gemmBlocks, 256, 0, stream>>>(x, Wg[1], xw);
    scatter_kernel<<<EE / 4, 256, 0, stream>>>(xw, srcI, dstI, nrm, agg);
    gate_gemm<1><<<gemmBlocks, 256, 0, stream>>>(agg, bg[1], Hprev, nullptr, nullptr,
                                                 Lg[1], lg[1], Rb);
    // ---- gate h + GRU combine -> H_new ----
    hipMemsetAsync(agg, 0, (size_t)R64 * sizeof(float), stream);
    xw_gemm<<<gemmBlocks, 256, 0, stream>>>(x, Wg[2], xw);
    scatter_kernel<<<EE / 4, 256, 0, stream>>>(xw, srcI, dstI, nrm, agg);
    gate_gemm<2><<<gemmBlocks, 256, 0, stream>>>(agg, bg[2], Hprev, Rb, Zb,
                                                 Lg[2], lg[2], Hnew);

    // ---- global attention pooling + head ----
    attn_gate_kernel<<<(RTOT * 32 + 255) / 256, 256, 0, stream>>>(Hnew, gw, gb, gate);
    softmax_stats<<<BB, 256, 0, stream>>>(gate, gstat);
    pool_kernel<<<BB, 256, 0, stream>>>(Hnew, gate, gstat, ypool);
    head_kernel<<<1, 256, 0, stream>>>(ypool, W1, b1, W2, b2, y_out);
}
